// GraphAttentionLayerOriginal_7739531067612
// MI455X (gfx1250) — matrix operand order repaired and checked by reading, not compiled, not run on hardware
//
#include <hip/hip_runtime.h>
#include <hip/hip_bf16.h>
#include <stdint.h>

#define B_    8
#define N_    2048
#define DIN_  256
#define D_    256
#define DFF_  1024
#define M_    (B_ * N_)     // 16384 rows total
#define ALPHA_ 0.2f
#define EPS_   1e-5f

typedef unsigned short u16;
typedef __attribute__((ext_vector_type(16))) __bf16 v16bf;
typedef __attribute__((ext_vector_type(8)))  float  v8f;
typedef __attribute__((ext_vector_type(4)))  unsigned int v4u;
typedef __attribute__((ext_vector_type(8)))  int v8i;
typedef __attribute__((ext_vector_type(4)))  int v4i;

// ---------------------------------------------------------------- helpers
__device__ __forceinline__ u16 f2bf(float f) {
    union { float f; unsigned u; } x; x.f = f;
    unsigned u = x.u;
    unsigned r = (u + 0x7FFFu + ((u >> 16) & 1u)) >> 16;  // RTNE
    return (u16)r;
}

__global__ __launch_bounds__(256)
void cvt_bf16_kernel(const float* __restrict__ in, u16* __restrict__ out, int n) {
    int i = blockIdx.x * 256 + threadIdx.x;
    if (i < n) out[i] = f2bf(in[i]);
}

// ---------------------------------------------------------------- TDM issue
// Load a 32(k) x 64(n) bf16 tile of a row-major [Nc x K] matrix into LDS.
// D# packing per CDNA5 ISA (2D tensor,
// data_size=2B, dim0 = contiguous K axis, dim0_stride = K elements).
__device__ __forceinline__ void tdm_load_w_tile(const u16* gptr, unsigned lds_off,
                                                int K, int Nc) {
    unsigned long long ga = (unsigned long long)(size_t)gptr;
    v4u g0;
    g0.x = 1u;                                           // count=1, is_restore=0
    g0.y = lds_off;                                      // lds_addr
    g0.z = (unsigned)ga;                                 // global_addr[31:0]
    g0.w = (unsigned)((ga >> 32) & 0x01FFFFFFull)        // global_addr[56:32]
         | (2u << 30);                                   // type=2 ("image")
    v8i g1;
    g1[0] = 0x00010000;                                  // wg_mask=0, data_size=2B
    g1[1] = (int)(((unsigned)K & 0xFFFFu) << 16);        // tensor_dim0 lo16
    g1[2] = (int)(((unsigned)K >> 16) |
                  (((unsigned)Nc & 0xFFFFu) << 16));     // dim0 hi16 | dim1 lo16
    g1[3] = (int)(((unsigned)Nc >> 16) | (32u << 16));   // dim1 hi16 | tile_dim0=32
    g1[4] = 64;                                          // tile_dim1=64, tile_dim2=0
    g1[5] = K;                                           // tensor_dim0_stride lo32
    g1[6] = 0;                                           // stride hi | dim1_stride lo
    g1[7] = 0;
    v4i z4 = {0, 0, 0, 0};
    v8i z8 = {0, 0, 0, 0, 0, 0, 0, 0};
    __builtin_amdgcn_tensor_load_to_lds(g0, g1, z4, z4, z8, 0);
}

// ---------------------------------------------------------------- WMMA GEMM
// C[M,Nc] = A[M,K] * W[Nc,K]^T + bias, optional ReLU, out fp32 or bf16.
// Block = 128 threads (4 waves) computing a 128(M) x 64(N) tile.
// Per wave: 2 M-subtiles x 4 N-subtiles = 8 accumulators, 8 WMMA per K-step
// reusing 4 B fragments staged in LDS by TDM (double-buffered).
// K-loop unrolled 2x with ping-pong A fragments (no register copies).
union FragBF { v16bf v; uint4 q[2]; };

template<int RELU, int OUT_BF16>
__global__ __launch_bounds__(128)
void gemm_bf16_wmma(const u16* __restrict__ A, const u16* __restrict__ W,
                    const float* __restrict__ bias, void* __restrict__ outp,
                    int M, int Nc, int K)
{
    __shared__ __align__(16) u16 bsm[2][64 * 32];   // 2 x 4KB weight tiles

    const int lane = threadIdx.x & 31;
    const int wid  = threadIdx.x >> 5;
    const int half = lane >> 4;
    const int l16  = lane & 15;
    const int mb   = blockIdx.x * 128 + wid * 32;       // wave's first row
    const int nt0  = blockIdx.y * 64;                   // first output column

    v8f acc[8] = {};

    const u16* arow0 = A + (size_t)(mb + l16) * K;        // M-subtile 0
    const u16* arow1 = A + (size_t)(mb + 16 + l16) * K;   // M-subtile 1

    // A 16x32 bf16 fragments (ISA layout): lanes 0-15 K=[0..7]/[16..23],
    // lanes 16-31 K=[8..15]/[24..31].  e = even K-step, o = odd K-step.
    FragBF a0e, a1e, a0o, a1o;
    a0e.q[0] = *(const uint4*)(arow0 + 8 * half);
    a0e.q[1] = *(const uint4*)(arow0 + 16 + 8 * half);
    a1e.q[0] = *(const uint4*)(arow1 + 8 * half);
    a1e.q[1] = *(const uint4*)(arow1 + 16 + 8 * half);
    a0o.q[0] = *(const uint4*)(arow0 + 32 + 8 * half);
    a0o.q[1] = *(const uint4*)(arow0 + 48 + 8 * half);
    a1o.q[0] = *(const uint4*)(arow1 + 32 + 8 * half);
    a1o.q[1] = *(const uint4*)(arow1 + 48 + 8 * half);

    const unsigned lds0 = (unsigned)(size_t)&bsm[0][0];
    const unsigned lds1 = (unsigned)(size_t)&bsm[1][0];
    if (wid == 0) tdm_load_w_tile(W + (size_t)nt0 * K, lds0, K, Nc);

    // load 4 B frags from one LDS buffer, fire 8 WMMAs against two A frags
    auto do_step = [&](const FragBF& f0, const FragBF& f1, int bufi) {
        const int base = l16 * 32 + 8 * half;
        FragBF b0, b1, b2, b3;
        b0.q[0] = *(const uint4*)&bsm[bufi][base];
        b0.q[1] = *(const uint4*)&bsm[bufi][base + 16];
        b1.q[0] = *(const uint4*)&bsm[bufi][base + 512];
        b1.q[1] = *(const uint4*)&bsm[bufi][base + 528];
        b2.q[0] = *(const uint4*)&bsm[bufi][base + 1024];
        b2.q[1] = *(const uint4*)&bsm[bufi][base + 1040];
        b3.q[0] = *(const uint4*)&bsm[bufi][base + 1536];
        b3.q[1] = *(const uint4*)&bsm[bufi][base + 1552];
        acc[0] = __builtin_amdgcn_wmma_f32_16x16x32_bf16(false, f0.v, false, b0.v, (short)0, acc[0], false, false);
        acc[1] = __builtin_amdgcn_wmma_f32_16x16x32_bf16(false, f0.v, false, b1.v, (short)0, acc[1], false, false);
        acc[2] = __builtin_amdgcn_wmma_f32_16x16x32_bf16(false, f0.v, false, b2.v, (short)0, acc[2], false, false);
        acc[3] = __builtin_amdgcn_wmma_f32_16x16x32_bf16(false, f0.v, false, b3.v, (short)0, acc[3], false, false);
        acc[4] = __builtin_amdgcn_wmma_f32_16x16x32_bf16(false, f1.v, false, b0.v, (short)0, acc[4], false, false);
        acc[5] = __builtin_amdgcn_wmma_f32_16x16x32_bf16(false, f1.v, false, b1.v, (short)0, acc[5], false, false);
        acc[6] = __builtin_amdgcn_wmma_f32_16x16x32_bf16(false, f1.v, false, b2.v, (short)0, acc[6], false, false);
        acc[7] = __builtin_amdgcn_wmma_f32_16x16x32_bf16(false, f1.v, false, b3.v, (short)0, acc[7], false, false);
    };

    const int nsteps = K >> 5;   // always even here (8 or 32)
    for (int step = 0; step < nsteps; step += 2) {
        // ---- even K-step: consume buf0 ----
        __syncthreads();   // everyone done reading buf1 (prev odd step)
        if (wid == 0) __builtin_amdgcn_s_wait_tensorcnt(0);
        __syncthreads();   // buf0 visible to all waves
        if (wid == 0 && step + 1 < nsteps)
            tdm_load_w_tile(W + (size_t)nt0 * K + (step + 1) * 32, lds1, K, Nc);
        do_step(a0e, a1e, 0);
        if (step + 2 < nsteps) {           // prefetch A for next even step
            const u16* p0 = arow0 + (step + 2) * 32;
            const u16* p1 = arow1 + (step + 2) * 32;
            a0e.q[0] = *(const uint4*)(p0 + 8 * half);
            a0e.q[1] = *(const uint4*)(p0 + 16 + 8 * half);
            a1e.q[0] = *(const uint4*)(p1 + 8 * half);
            a1e.q[1] = *(const uint4*)(p1 + 16 + 8 * half);
        }
        // ---- odd K-step: consume buf1 ----
        __syncthreads();   // everyone done reading buf0
        if (wid == 0) __builtin_amdgcn_s_wait_tensorcnt(0);
        __syncthreads();   // buf1 visible to all waves
        if (wid == 0 && step + 2 < nsteps)
            tdm_load_w_tile(W + (size_t)nt0 * K + (step + 2) * 32, lds0, K, Nc);
        do_step(a0o, a1o, 1);
        if (step + 3 < nsteps) {           // prefetch A for next odd step
            const u16* p0 = arow0 + (step + 3) * 32;
            const u16* p1 = arow1 + (step + 3) * 32;
            a0o.q[0] = *(const uint4*)(p0 + 8 * half);
            a0o.q[1] = *(const uint4*)(p0 + 16 + 8 * half);
            a1o.q[0] = *(const uint4*)(p1 + 8 * half);
            a1o.q[1] = *(const uint4*)(p1 + 16 + 8 * half);
        }
    }

    // C/D layout: VGPR e -> M = e + 8*half, N = l16
#pragma unroll
    for (int g = 0; g < 2; ++g) {
#pragma unroll
        for (int s = 0; s < 4; ++s) {
            int col = nt0 + s * 16 + l16;
            float bv = bias[col];
#pragma unroll
            for (int e = 0; e < 8; ++e) {
                int m = g * 16 + e + 8 * half;
                float v = acc[g * 4 + s][e] + bv;
                if (RELU) v = v > 0.f ? v : 0.f;
                size_t off = (size_t)(mb + m) * Nc + col;
                if (OUT_BF16) ((u16*)outp)[off] = f2bf(v);
                else          ((float*)outp)[off] = v;
            }
        }
    }
}

// ---------------------------------------------------------------- s1 / s2
__global__ __launch_bounds__(256)
void s1s2_kernel(const float* __restrict__ Wh, const float* __restrict__ a_w,
                 float* __restrict__ s1, float* __restrict__ s2)
{
    int wid = threadIdx.x >> 5, lane = threadIdx.x & 31;
    int row = blockIdx.x * 8 + wid;
    const float* wr = Wh + (size_t)row * D_;
    float acc1 = 0.f, acc2 = 0.f;
#pragma unroll
    for (int t = 0; t < D_ / 32; ++t) {
        int c = lane + 32 * t;
        float v = wr[c];
        acc1 += v * a_w[c];
        acc2 += v * a_w[D_ + c];
    }
#pragma unroll
    for (int o = 16; o >= 1; o >>= 1) {
        acc1 += __shfl_xor(acc1, o, 32);
        acc2 += __shfl_xor(acc2, o, 32);
    }
    if (lane == 0) { s1[row] = acc1; s2[row] = acc2; }
}

// ---------------------------------------------------------------- attention
// Only B*B = 64 distinct attention rows exist. One block per (b, r=i%B).
__global__ __launch_bounds__(256)
void attn_kernel(const float* __restrict__ Wh, const float* __restrict__ s1,
                 const float* __restrict__ s2, const float* __restrict__ a_b,
                 float* __restrict__ hp)
{
    __shared__ float p[N_];
    __shared__ float red[256];
    const int b  = blockIdx.x >> 3;
    const int r  = blockIdx.x & 7;
    const int rr = (int)(((long long)b * N_ + r) % B_);   // idx[b, i] for i%B == r
    const int tid = threadIdx.x;
    const float ab = a_b[0];
    const float* s1b = s1 + (size_t)b * N_;
    const float* s2r = s2 + (size_t)rr * N_;

    float lmax = -3.4e38f;
    for (int j = tid; j < N_; j += 256) {
        float t = s1b[j] + s2r[j] + ab;
        float e = t > 0.f ? t : ALPHA_ * t;
        p[j] = e;
        lmax = fmaxf(lmax, e);
    }
    red[tid] = lmax; __syncthreads();
    for (int o = 128; o >= 1; o >>= 1) {
        if (tid < o) red[tid] = fmaxf(red[tid], red[tid + o]);
        __syncthreads();
    }
    const float smax = red[0];
    __syncthreads();

    float lsum = 0.f;
    for (int j = tid; j < N_; j += 256) {
        float pe = __expf(p[j] - smax);
        p[j] = pe;
        lsum += pe;
    }
    red[tid] = lsum; __syncthreads();
    for (int o = 128; o >= 1; o >>= 1) {
        if (tid < o) red[tid] += red[tid + o];
        __syncthreads();
    }
    const float inv = 1.f / red[0];

    // weighted sum: each thread owns output dim d = tid
    float acc = 0.f;
    const float* whb = Wh + (size_t)b * N_ * D_ + tid;
    for (int j = 0; j < N_; j += 4) {
        acc += p[j + 0] * whb[(size_t)(j + 0) * D_];
        acc += p[j + 1] * whb[(size_t)(j + 1) * D_];
        acc += p[j + 2] * whb[(size_t)(j + 2) * D_];
        acc += p[j + 3] * whb[(size_t)(j + 3) * D_];
    }
    hp[(size_t)blockIdx.x * D_ + tid] = acc * inv;
}

// ---------------------------------------------------------------- LayerNorms
__global__ __launch_bounds__(256)
void ln1_kernel(const float* __restrict__ h, const float* __restrict__ hp,
                const float* __restrict__ g1, const float* __restrict__ b1,
                float* __restrict__ x, u16* __restrict__ xbf)
{
    int wid = threadIdx.x >> 5, lane = threadIdx.x & 31;
    int row = blockIdx.x * 8 + wid;
    int b = row / N_;
    int i = row % N_;
    const float* hrow  = h  + (size_t)row * D_;
    const float* hprow = hp + (size_t)(b * B_ + (i % B_)) * D_;
    float vals[8];
    float sum = 0.f;
#pragma unroll
    for (int t = 0; t < 8; ++t) {
        int c = lane + 32 * t;
        float v = hrow[c] + hprow[c];
        vals[t] = v; sum += v;
    }
#pragma unroll
    for (int o = 16; o >= 1; o >>= 1) sum += __shfl_xor(sum, o, 32);
    float mean = sum * (1.f / D_);
    float var = 0.f;
#pragma unroll
    for (int t = 0; t < 8; ++t) { float d = vals[t] - mean; var += d * d; }
#pragma unroll
    for (int o = 16; o >= 1; o >>= 1) var += __shfl_xor(var, o, 32);
    float rstd = rsqrtf(var * (1.f / D_) + EPS_);
    float* xrow = x + (size_t)row * D_;
    u16* xbrow  = xbf + (size_t)row * D_;
#pragma unroll
    for (int t = 0; t < 8; ++t) {
        int c = lane + 32 * t;
        float v = (vals[t] - mean) * rstd * g1[c] + b1[c];
        xrow[c] = v;
        xbrow[c] = f2bf(v);
    }
}

__global__ __launch_bounds__(256)
void ln2_kernel(const float* __restrict__ x, const float* __restrict__ y2,
                const float* __restrict__ g2, const float* __restrict__ b2,
                float* __restrict__ out)
{
    int wid = threadIdx.x >> 5, lane = threadIdx.x & 31;
    int row = blockIdx.x * 8 + wid;
    const float* xrow = x  + (size_t)row * D_;
    const float* yrow = y2 + (size_t)row * D_;
    float vals[8];
    float sum = 0.f;
#pragma unroll
    for (int t = 0; t < 8; ++t) {
        int c = lane + 32 * t;
        float v = xrow[c] + yrow[c];
        vals[t] = v; sum += v;
    }
#pragma unroll
    for (int o = 16; o >= 1; o >>= 1) sum += __shfl_xor(sum, o, 32);
    float mean = sum * (1.f / D_);
    float var = 0.f;
#pragma unroll
    for (int t = 0; t < 8; ++t) { float d = vals[t] - mean; var += d * d; }
#pragma unroll
    for (int o = 16; o >= 1; o >>= 1) var += __shfl_xor(var, o, 32);
    float rstd = rsqrtf(var * (1.f / D_) + EPS_);
    float* orow = out + (size_t)row * D_;
#pragma unroll
    for (int t = 0; t < 8; ++t) {
        int c = lane + 32 * t;
        orow[c] = (vals[t] - mean) * rstd * g2[c] + b2[c];
    }
}

// ---------------------------------------------------------------- launch
extern "C" void kernel_launch(void* const* d_in, const int* in_sizes, int n_in,
                              void* d_out, int out_size, void* d_ws, size_t ws_size,
                              hipStream_t stream)
{
    const float* h    = (const float*)d_in[0];
    // d_in[1] = adj: unused by the reference computation (all-ones, never applied)
    const float* W_w  = (const float*)d_in[2];
    const float* W_b  = (const float*)d_in[3];
    const float* a_w  = (const float*)d_in[4];
    const float* a_b  = (const float*)d_in[5];
    const float* c1_w = (const float*)d_in[6];
    const float* c1_b = (const float*)d_in[7];
    const float* c2_w = (const float*)d_in[8];
    const float* c2_b = (const float*)d_in[9];
    const float* g1   = (const float*)d_in[10];
    const float* b1   = (const float*)d_in[11];
    const float* g2   = (const float*)d_in[12];
    const float* b2   = (const float*)d_in[13];
    (void)in_sizes; (void)n_in; (void)out_size; (void)ws_size;

    char* ws = (char*)d_ws;
    size_t off = 0;
    auto carve = [&](size_t bytes) -> void* {
        void* p = ws + off;
        off = (off + bytes + 255) & ~(size_t)255;
        return p;
    };

    u16*   h_bf  = (u16*)  carve((size_t)M_ * DIN_ * 2);      //  8 MB
    u16*   w1_bf = (u16*)  carve((size_t)D_ * DIN_ * 2);      // 128 KB
    u16*   c1_bf = (u16*)  carve((size_t)DFF_ * D_ * 2);      // 512 KB
    u16*   c2_bf = (u16*)  carve((size_t)D_ * DFF_ * 2);      // 512 KB
    float* Wh    = (float*)carve((size_t)M_ * D_ * 4);        // 16 MB (reused as y2)
    float* s1    = (float*)carve((size_t)M_ * 4);
    float* s2    = (float*)carve((size_t)M_ * 4);
    float* hp    = (float*)carve((size_t)B_ * B_ * D_ * 4);
    float* x     = (float*)carve((size_t)M_ * D_ * 4);        // 16 MB
    u16*   x_bf  = (u16*)  carve((size_t)M_ * D_ * 2);        //  8 MB
    u16*   y1_bf = (u16*)  carve((size_t)M_ * DFF_ * 2);      // 32 MB
    float* y2    = Wh;  // Wh is dead after attn_kernel

    // 1) bf16 conversions
    cvt_bf16_kernel<<<(M_ * DIN_ + 255) / 256, 256, 0, stream>>>(h, h_bf, M_ * DIN_);
    cvt_bf16_kernel<<<(D_ * DIN_ + 255) / 256, 256, 0, stream>>>(W_w, w1_bf, D_ * DIN_);
    cvt_bf16_kernel<<<(DFF_ * D_ + 255) / 256, 256, 0, stream>>>(c1_w, c1_bf, DFF_ * D_);
    cvt_bf16_kernel<<<(D_ * DFF_ + 255) / 256, 256, 0, stream>>>(c2_w, c2_bf, D_ * DFF_);

    // 2) Wh = h @ W_w^T + W_b   (M=16384, Nc=256, K=256)
    gemm_bf16_wmma<0, 0><<<dim3(M_ / 128, D_ / 64), 128, 0, stream>>>(
        h_bf, w1_bf, W_b, Wh, M_, D_, DIN_);

    // 3) s1 = Wh@a1, s2 = Wh@a2
    s1s2_kernel<<<M_ / 8, 256, 0, stream>>>(Wh, a_w, s1, s2);

    // 4) 64 distinct attention rows -> hp[B][B][D]
    attn_kernel<<<B_ * B_, 256, 0, stream>>>(Wh, s1, s2, a_b, hp);

    // 5) x = LN(h_prime + h); also bf16 copy for FFN
    ln1_kernel<<<M_ / 8, 256, 0, stream>>>(h, hp, g1, b1, x, x_bf);

    // 6) y1 = relu(x @ c1_w^T + c1_b)   (Nc=1024, K=256), bf16 out
    gemm_bf16_wmma<1, 1><<<dim3(M_ / 128, DFF_ / 64), 128, 0, stream>>>(
        x_bf, c1_bf, c1_b, y1_bf, M_, DFF_, D_);

    // 7) y2 = y1 @ c2_w^T + c2_b   (Nc=256, K=1024), fp32 out
    gemm_bf16_wmma<0, 0><<<dim3(M_ / 128, D_ / 64), 128, 0, stream>>>(
        y1_bf, c2_bf, c2_b, y2, M_, D_, DFF_);

    // 8) out = LN(x + y2) * g2 + b2
    ln2_kernel<<<M_ / 8, 256, 0, stream>>>(x, y2, g2, b2, (float*)d_out);
}
